// PaiNN_33913061769770
// MI455X (gfx1250) — matrix operand order repaired and checked by reading, not compiled, not run on hardware
//
#include <hip/hip_runtime.h>
#include <hip/hip_bf16.h>
#include <math.h>

// ---------------------------------------------------------------------------
// PaiNN forward for MI455X (gfx1250, wave32, WMMA).
//
// All GEMMs run on v_wmma_f32_16x16x32_bf16 (bf16 A/B, f32 accumulate).
// The big edge-filter GEMM ([E,50pad64] @ [50,384] per layer) is fused into
// the edge-message kernel: the layer's packed W_filt slice (48 KB bf16) and
// bias live in LDS, so the 737 MB "filters" tensor of the reference is never
// materialized. Segment sums use global_atomic_add_f32 into L2-resident
// accumulators.
//
// Register-fragment layouts (CDNA5 ISA 7.12.2):
//   A (16-bit, 16x32): lane(0-15)=row M, VGPR0-3 hold K= 0..7 (lo)/8..15 (hi),
//                      VGPR4-7 hold K=16..23 (lo)/24..31 (hi). -> two b128 loads.
//   C/D (f32 16x16):   VGPR v: row = v + 8*(lane>=16), col = lane&15.
//   B (16-bit, 32x16): packed by pack_b_kernel as: col = lane&15,
//                      K = 2v + 8*(lane>=16) + 8*(v>=4) within a 32-deep k-step
//                      (same K order as the A fragment).
//                      Pack and load use the same convention.
// ---------------------------------------------------------------------------

#define N_NODES 10000
#define N_EDGES 160000
#define FDIM    128
#define NLAYERS 3
#define NRBF    50
#define INDIM   100
#define CUTOFF  5.0f
#define PI_F    3.14159265358979323846f

typedef __attribute__((ext_vector_type(16))) __bf16       v16bf;
typedef __attribute__((ext_vector_type(8)))  float        v8f;
typedef __attribute__((ext_vector_type(4)))  unsigned int v4u;

// ---------------- bf16 helpers (round-to-nearest-even) ----------------------
static __device__ __forceinline__ unsigned short f2bf_bits(float f) {
    union { float f; unsigned u; } v; v.f = f;
    unsigned u = v.u;
    unsigned r = u + 0x7FFFu + ((u >> 16) & 1u);
    return (unsigned short)(r >> 16);
}
static __device__ __forceinline__ __bf16 f2bf(float f) {
    union { unsigned short s; __bf16 b; } c; c.s = f2bf_bits(f); return c.b;
}

// ---------------- fragment loaders ------------------------------------------
// A fragment: row-major bf16 A, row stride = kpad elements, one 16-row tile.
static __device__ __forceinline__ v16bf load_a_frag(const __bf16* A, int row,
                                                    int kpad, int kstep) {
    const int lane = threadIdx.x & 31;
    const int hi   = lane >> 4;
    const __bf16* p = A + (size_t)row * kpad + kstep * 32 + hi * 8;
    union { v16bf bf; v4u u[2]; } r;
    r.u[0] = *(const v4u*)(p);
    r.u[1] = *(const v4u*)(p + 16);
    return r.bf;
}
// B fragment from packed LDS image: 256 dwords per fragment, 8 dwords per lane.
static __device__ __forceinline__ v16bf load_b_frag(const unsigned* ldsB, int frag) {
    const int lane = threadIdx.x & 31;
    const unsigned* p = ldsB + frag * 256 + lane * 8;
    union { v16bf bf; v4u u[2]; } r;
    r.u[0] = *(const v4u*)(p);
    r.u[1] = *(const v4u*)(p + 4);
    return r.bf;
}

static __device__ __forceinline__ float silu_f(float x) {
    return x * (1.0f / (1.0f + __expf(-x)));
}

// ---------------- weight packing: f32 [K x Nn] -> bf16 B fragments ----------
__global__ void pack_b_kernel(const float* __restrict__ W, int ldw, int colOff,
                              int Ksrc, int Kpad, int Nn, unsigned* __restrict__ out) {
    const int frag = blockIdx.x;          // one fragment per block (32 lanes)
    const int lane = threadIdx.x;
    const int ksteps = Kpad >> 5;
    const int nt = frag / ksteps;
    const int s  = frag % ksteps;
    const int lo = lane & 15, hi = lane >> 4;
    const int col = nt * 16 + lo;
    unsigned* dst = out + (size_t)frag * 256 + lane * 8;
#pragma unroll
    for (int v = 0; v < 8; ++v) {
        const int k0 = s * 32 + hi * 8 + 2 * v + (v >= 4 ? 8 : 0);
        unsigned short b0 = 0, b1 = 0;
        if (k0     < Ksrc) b0 = f2bf_bits(W[(size_t)k0 * ldw + colOff + col]);
        if (k0 + 1 < Ksrc) b1 = f2bf_bits(W[(size_t)(k0 + 1) * ldw + colOff + col]);
        dst[v] = (unsigned)b0 | ((unsigned)b1 << 16);
    }
}

// ---------------- h (f32 [N,100]) -> bf16 [N,128] zero-padded ---------------
__global__ void convert_h_kernel(const float* __restrict__ h, __bf16* __restrict__ hb,
                                 int total) {
    const int i = blockIdx.x * blockDim.x + threadIdx.x;
    if (i >= total) return;
    const int r = i >> 7, c = i & 127;
    const float v = (c < INDIM) ? h[(size_t)r * INDIM + c] : 0.0f;
    hb[i] = f2bf(v);
}

// ---------------- per-edge geometry + Gaussian RBF (bf16, K padded to 64) ---
__global__ void edge_geom_kernel(const float* __restrict__ x,
                                 const int* __restrict__ idx_i,
                                 const int* __restrict__ idx_j,
                                 float* __restrict__ dir, float* __restrict__ fcut,
                                 __bf16* __restrict__ phi) {
    const int e = blockIdx.x * blockDim.x + threadIdx.x;
    if (e >= N_EDGES) return;
    const int ii = idx_i[e], jj = idx_j[e];
    const float rx = x[jj * 3 + 0] - x[ii * 3 + 0];
    const float ry = x[jj * 3 + 1] - x[ii * 3 + 1];
    const float rz = x[jj * 3 + 2] - x[ii * 3 + 2];
    const float d  = sqrtf(rx * rx + ry * ry + rz * rz);
    const float inv = 1.0f / d;
    dir[e * 3 + 0] = rx * inv;
    dir[e * 3 + 1] = ry * inv;
    dir[e * 3 + 2] = rz * inv;
    fcut[e] = (d < CUTOFF) ? 0.5f * (__cosf(d * (PI_F / CUTOFF)) + 1.0f) : 0.0f;
    const float delta = CUTOFF / (float)(NRBF - 1);
    const float coeff = -0.5f / (delta * delta);
    __bf16* pr = phi + (size_t)e * 64;
#pragma unroll
    for (int k = 0; k < 64; ++k) {
        float v = 0.0f;
        if (k < NRBF) { const float t = d - (float)k * delta; v = __expf(coeff * t * t); }
        pr[k] = f2bf(v);
    }
}

// ---------------- generic bf16 WMMA GEMM: C = act(A @ B + bias) -------------
// A: bf16 row-major [M x 32*KSTEPS]; B: packed fragments (staged in LDS).
// All configuration is compile-time so the inner loop is a branch-free
// ds_load_b128 x2 -> v_wmma sequence per k-step.
template <int KSTEPS, int ACT, bool HAS_BIAS, bool F32OUT, bool BF16OUT>
__global__ void __launch_bounds__(128) gemm_bf16_kernel(
        const __bf16* __restrict__ A, const unsigned* __restrict__ Bpack,
        const float* __restrict__ bias, float* __restrict__ Cf32,
        __bf16* __restrict__ Cbf16, int M, int Nn) {
    extern __shared__ __align__(16) unsigned lds[];
    constexpr int KPAD = KSTEPS * 32;
    const int ntiles = Nn >> 4;
    const int nbw    = ntiles * KSTEPS * 256;       // B image dwords (mult of 256)
    float* ldsBias = (float*)(lds + nbw);
    const int tid = threadIdx.x;
    for (int i = tid; i < (nbw >> 2); i += blockDim.x)
        ((v4u*)lds)[i] = ((const v4u*)Bpack)[i];
    if (HAS_BIAS)
        for (int i = tid; i < Nn; i += blockDim.x) ldsBias[i] = bias[i];
    __syncthreads();

    const int wave = tid >> 5, lane = tid & 31, lo = lane & 15, hi = lane >> 4;
    const int m0 = (blockIdx.x * 4 + wave) * 16;
    if (m0 >= M) return;
    const int arow = m0 + lo;

    v16bf afr[KSTEPS];
#pragma unroll
    for (int s = 0; s < KSTEPS; ++s) afr[s] = load_a_frag(A, arow, KPAD, s);

    for (int nt = 0; nt < ntiles; ++nt) {
        const int col = nt * 16 + lo;
        const float b = HAS_BIAS ? ldsBias[col] : 0.0f;
        v8f acc = { b, b, b, b, b, b, b, b };
#pragma unroll
        for (int s = 0; s < KSTEPS; ++s) {
            v16bf bb = load_b_frag(lds, nt * KSTEPS + s);
            acc = __builtin_amdgcn_wmma_f32_16x16x32_bf16(
                      false, afr[s], false, bb, (short)0, acc, false, false);
        }
#pragma unroll
        for (int v = 0; v < 8; ++v) {
            const int r = m0 + v + 8 * hi;
            float val = acc[v];
            if (ACT == 1) val = silu_f(val);
            if (F32OUT)  Cf32[(size_t)r * Nn + col]  = val;
            if (BF16OUT) Cbf16[(size_t)r * Nn + col] = f2bf(val);
        }
    }
}

// ---------------- fused edge-message kernel ---------------------------------
// Per 16-edge wave tile: WMMA computes (phi @ Wfilt_l + bfilt_l) for all 384
// filter columns (weights in LDS), epilogue applies fcut and the xc[idx_j]
// gather, tile goes to LDS, then dq / dmu are scatter-added with f32 atomics.
#define EDGE_WAVES 4
__global__ void __launch_bounds__(128) edge_msg_kernel(
        const __bf16* __restrict__ phi, const float* __restrict__ fcut,
        const float* __restrict__ dir, const int* __restrict__ idx_i,
        const int* __restrict__ idx_j, const float* __restrict__ xc,
        const float* __restrict__ mu_in, const unsigned* __restrict__ WfiltP,
        const float* __restrict__ bfilt_l, float* __restrict__ dq_buf,
        float* __restrict__ dmu_buf) {
    extern __shared__ __align__(16) unsigned lds[];
    unsigned* ldsB   = lds;                       // 24 ntiles * 2 ksteps * 256 dwords
    float*    ldsBia = (float*)(lds + 12288);     // 384 floats
    float*    tiles  = (float*)(lds + 12288 + 384);
    const int tid = threadIdx.x;
    for (int i = tid; i < (12288 >> 2); i += blockDim.x)
        ((v4u*)ldsB)[i] = ((const v4u*)WfiltP)[i];
    for (int i = tid; i < 384; i += blockDim.x) ldsBia[i] = bfilt_l[i];
    __syncthreads();

    const int wave = tid >> 5, lane = tid & 31, lo = lane & 15, hi = lane >> 4;
    const int ebase = (blockIdx.x * EDGE_WAVES + wave) * 16;
    float* tile = tiles + wave * (16 * 384);

    float fc[8]; int jj8[8];
#pragma unroll
    for (int v = 0; v < 8; ++v) {
        const int e = ebase + v + 8 * hi;
        fc[v] = fcut[e]; jj8[v] = idx_j[e];
    }
    const int arow = ebase + lo;
    v16bf a0 = load_a_frag(phi, arow, 64, 0);
    v16bf a1 = load_a_frag(phi, arow, 64, 1);

    for (int nt = 0; nt < 24; ++nt) {
        const int col = nt * 16 + lo;
        const float b = ldsBia[col];
        v8f acc = { b, b, b, b, b, b, b, b };
        v16bf b0 = load_b_frag(ldsB, nt * 2 + 0);
        acc = __builtin_amdgcn_wmma_f32_16x16x32_bf16(false, a0, false, b0,
                                                      (short)0, acc, false, false);
        v16bf b1 = load_b_frag(ldsB, nt * 2 + 1);
        acc = __builtin_amdgcn_wmma_f32_16x16x32_bf16(false, a1, false, b1,
                                                      (short)0, acc, false, false);
#pragma unroll
        for (int v = 0; v < 8; ++v) {
            const int r = v + 8 * hi;
            const float xe = acc[v] * fc[v] * xc[(size_t)jj8[v] * 384 + col];
            tile[r * 384 + col] = xe;
        }
    }
    __syncthreads();

    // scatter phase: each wave handles its own 16 edges
    for (int e = 0; e < 16; ++e) {
        const int eg = ebase + e;
        const int ii = idx_i[eg], jv = idx_j[eg];
        const float d0 = dir[eg * 3 + 0], d1 = dir[eg * 3 + 1], d2 = dir[eg * 3 + 2];
        const float* muj = mu_in + (size_t)jv * 384;
#pragma unroll
        for (int cc = 0; cc < 4; ++cc) {
            const int c = lane + cc * 32;
            atomicAdd(&dq_buf[(size_t)ii * 128 + c], tile[e * 384 + c]);
            const float dR = tile[e * 384 + 128 + c];
            const float dM = tile[e * 384 + 256 + c];
            atomicAdd(&dmu_buf[(size_t)ii * 384 + c],       dR * d0 + dM * muj[c]);
            atomicAdd(&dmu_buf[(size_t)ii * 384 + 128 + c], dR * d1 + dM * muj[128 + c]);
            atomicAdd(&dmu_buf[(size_t)ii * 384 + 256 + c], dR * d2 + dM * muj[256 + c]);
        }
    }
}

// ---------------- node update: q += dq; mu += dmu (+ bf16 mirror of mu) -----
__global__ void node_update_kernel(float* __restrict__ q, float* __restrict__ mu,
                                   const float* __restrict__ dq,
                                   const float* __restrict__ dmu,
                                   __bf16* __restrict__ mu_bf, int qn, int mun) {
    const int i = blockIdx.x * blockDim.x + threadIdx.x;
    if (i < qn) {
        q[i] += dq[i];
    } else {
        const int j = i - qn;
        if (j < mun) {
            const float v = mu[j] + dmu[j];
            mu[j] = v;
            mu_bf[j] = f2bf(v);
        }
    }
}

// ---------------- mixing step 1: ||mu_V|| and ctx = [q, mu_Vn] (bf16) -------
__global__ void mix1_kernel(const float* __restrict__ q,
                            const float* __restrict__ mu_mix,
                            __bf16* __restrict__ ctx, int total) {
    const int i = blockIdx.x * blockDim.x + threadIdx.x;
    if (i >= total) return;
    const int n = i >> 7, c = i & 127;
    const float* mm = mu_mix + (size_t)n * 768;                 // [3][256]
    const float v0 = mm[c], v1 = mm[256 + c], v2 = mm[512 + c]; // mu_V
    const float vn = sqrtf(v0 * v0 + v1 * v1 + v2 * v2 + 1e-8f);
    ctx[(size_t)n * 256 + c]       = f2bf(q[i]);
    ctx[(size_t)n * 256 + 128 + c] = f2bf(vn);
}

// ---------------- mixing step 2: intra-atom updates -------------------------
__global__ void mix2_kernel(float* __restrict__ q, float* __restrict__ mu,
                            const float* __restrict__ xm,
                            const float* __restrict__ mu_mix,
                            __bf16* __restrict__ q_bf, int total) {
    const int i = blockIdx.x * blockDim.x + threadIdx.x;
    if (i >= total) return;
    const int n = i >> 7, c = i & 127;
    const float* xr = xm + (size_t)n * 384;
    const float* mm = mu_mix + (size_t)n * 768;
    // s = sum_axes(mu_V * mu_W); mu_W = mu_mix[:, :, 128:256]
    const float s = mm[c]       * mm[128 + c]
                  + mm[256 + c] * mm[384 + c]
                  + mm[512 + c] * mm[640 + c];
    const float qq = q[i] + xr[c] + xr[256 + c] * s;
    q[i] = qq;
    q_bf[i] = f2bf(qq);
    const float dmu1 = xr[128 + c];
    mu[(size_t)n * 384 + c]       += dmu1 * mm[128 + c];
    mu[(size_t)n * 384 + 128 + c] += dmu1 * mm[384 + c];
    mu[(size_t)n * 384 + 256 + c] += dmu1 * mm[640 + c];
}

// ---------------- host-side templated GEMM launcher -------------------------
template <int KSTEPS, int ACT, bool HAS_BIAS, bool F32OUT, bool BF16OUT>
static void launch_gemm(const __bf16* A, const unsigned* Bp, const float* bias,
                        float* Cf, __bf16* Cb, int M, int Nn, hipStream_t stream) {
    const int tiles  = (M + 15) / 16;
    const int blocks = (tiles + 3) / 4;
    const size_t smem = (size_t)KSTEPS * (Nn / 16) * 1024 + (size_t)Nn * 4;
    gemm_bf16_kernel<KSTEPS, ACT, HAS_BIAS, F32OUT, BF16OUT>
        <<<dim3(blocks), dim3(128), smem, stream>>>(A, Bp, bias, Cf, Cb, M, Nn);
}

// ---------------------------------------------------------------------------
extern "C" void kernel_launch(void* const* d_in, const int* in_sizes, int n_in,
                              void* d_out, int out_size, void* d_ws, size_t ws_size,
                              hipStream_t stream) {
    const float* x     = (const float*)d_in[0];
    const float* h     = (const float*)d_in[1];
    const float* Wemb  = (const float*)d_in[2];
    const float* bemb  = (const float*)d_in[3];
    const float* Wfilt = (const float*)d_in[4];
    const float* bfilt = (const float*)d_in[5];
    const float* Wint1 = (const float*)d_in[6];
    const float* bint1 = (const float*)d_in[7];
    const float* Wint2 = (const float*)d_in[8];
    const float* bint2 = (const float*)d_in[9];
    const float* Wmu   = (const float*)d_in[10];
    const float* Wmix1 = (const float*)d_in[11];
    const float* bmix1 = (const float*)d_in[12];
    const float* Wmix2 = (const float*)d_in[13];
    const float* bmix2 = (const float*)d_in[14];
    const int*   idx_i = (const int*)d_in[15];
    const int*   idx_j = (const int*)d_in[16];

    float* out = (float*)d_out;
    float* q   = out;                                 // [N,128] f32 (output 0)
    float* mu  = out + (size_t)N_NODES * FDIM;        // [N,3,128] f32 (output 1)

    // -------- workspace bump allocator (256B aligned) --------
    char* ws = (char*)d_ws;
    size_t off = 0;
    auto alloc = [&](size_t bytes) -> void* {
        off = (off + 255) & ~(size_t)255;
        void* p = ws + off;
        off += bytes;
        return p;
    };

    __bf16* h_bf    = (__bf16*)alloc((size_t)N_NODES * 128 * 2);
    __bf16* q_bf    = (__bf16*)alloc((size_t)N_NODES * 128 * 2);
    __bf16* t1_bf   = (__bf16*)alloc((size_t)N_NODES * 128 * 2);
    __bf16* t2_bf   = (__bf16*)alloc((size_t)N_NODES * 128 * 2);
    __bf16* ctx_bf  = (__bf16*)alloc((size_t)N_NODES * 256 * 2);
    __bf16* mu_bf   = (__bf16*)alloc((size_t)N_NODES * 384 * 2);
    __bf16* phi     = (__bf16*)alloc((size_t)N_EDGES * 64 * 2);
    float*  dir     = (float*)alloc((size_t)N_EDGES * 3 * 4);
    float*  fcut    = (float*)alloc((size_t)N_EDGES * 4);
    float*  xc      = (float*)alloc((size_t)N_NODES * 384 * 4);   // reused as xm
    float*  mu_mix  = (float*)alloc((size_t)N_NODES * 768 * 4);
    float*  dq_buf  = (float*)alloc((size_t)N_NODES * 128 * 4);
    float*  dmu_buf = (float*)alloc((size_t)N_NODES * 384 * 4);

    auto packB = [&](const float* W, int ldw, int colOff, int Ksrc, int Kpad,
                     int Nn) -> unsigned* {
        const int nfrag = (Kpad / 32) * (Nn / 16);
        unsigned* dst = (unsigned*)alloc((size_t)nfrag * 1024);
        pack_b_kernel<<<dim3(nfrag), dim3(32), 0, stream>>>(W, ldw, colOff, Ksrc,
                                                            Kpad, Nn, dst);
        return dst;
    };

    // -------- pack all weights to bf16 B-fragment images --------
    unsigned* WembP = packB(Wemb, 128, 0, INDIM, 128, 128);
    unsigned *Wint1P[NLAYERS], *Wint2P[NLAYERS], *WfiltP[NLAYERS];
    unsigned *WmuP[NLAYERS], *Wmix1P[NLAYERS], *Wmix2P[NLAYERS];
    for (int l = 0; l < NLAYERS; ++l) {
        Wint1P[l] = packB(Wint1 + (size_t)l * 128 * 128, 128, 0, 128, 128, 128);
        Wint2P[l] = packB(Wint2 + (size_t)l * 128 * 384, 384, 0, 128, 128, 384);
        WfiltP[l] = packB(Wfilt, NLAYERS * 384, l * 384, NRBF, 64, 384);
        WmuP[l]   = packB(Wmu   + (size_t)l * 128 * 256, 256, 0, 128, 128, 256);
        Wmix1P[l] = packB(Wmix1 + (size_t)l * 256 * 128, 128, 0, 256, 256, 128);
        Wmix2P[l] = packB(Wmix2 + (size_t)l * 128 * 384, 384, 0, 128, 128, 384);
    }

    // -------- precompute geometry / RBF; embed h --------
    convert_h_kernel<<<dim3((N_NODES * 128 + 255) / 256), dim3(256), 0, stream>>>(
        h, h_bf, N_NODES * 128);
    edge_geom_kernel<<<dim3((N_EDGES + 255) / 256), dim3(256), 0, stream>>>(
        x, idx_i, idx_j, dir, fcut, phi);

    // q0 = h@Wemb + b  (f32 into d_out, bf16 mirror for next GEMM)
    launch_gemm<4, 0, true, true, true>(h_bf, WembP, bemb, q, q_bf,
                                        N_NODES, 128, stream);
    hipMemsetAsync(mu, 0, (size_t)N_NODES * 384 * 4, stream);    // mu0 = 0

    const size_t edge_smem = 12288u * 4 + 384u * 4 +
                             (size_t)EDGE_WAVES * 16 * 384 * 4;  // ~146 KB

    for (int l = 0; l < NLAYERS; ++l) {
        // xc = silu(q@Wint1+b1) @ Wint2 + b2
        launch_gemm<4, 1, true, false, true>(q_bf, Wint1P[l],
                                             bint1 + (size_t)l * 128,
                                             nullptr, t1_bf, N_NODES, 128, stream);
        launch_gemm<4, 0, true, true, false>(t1_bf, Wint2P[l],
                                             bint2 + (size_t)l * 384,
                                             xc, nullptr, N_NODES, 384, stream);

        // edge messages (fused filter GEMM) -> dq_buf, dmu_buf
        hipMemsetAsync(dq_buf, 0, (size_t)N_NODES * 128 * 4, stream);
        hipMemsetAsync(dmu_buf, 0, (size_t)N_NODES * 384 * 4, stream);
        edge_msg_kernel<<<dim3(N_EDGES / (EDGE_WAVES * 16)), dim3(128), edge_smem,
                          stream>>>(phi, fcut, dir, idx_i, idx_j, xc, mu,
                                    WfiltP[l], bfilt + (size_t)l * 384,
                                    dq_buf, dmu_buf);

        // q += dq; mu += dmu (and mirror mu to bf16 for the Wmu GEMM)
        node_update_kernel<<<dim3((N_NODES * 512 + 255) / 256), dim3(256), 0,
                             stream>>>(q, mu, dq_buf, dmu_buf, mu_bf,
                                       N_NODES * 128, N_NODES * 384);

        // mixing
        launch_gemm<4, 0, false, true, false>(mu_bf, WmuP[l], nullptr,
                                              mu_mix, nullptr, 3 * N_NODES, 256,
                                              stream);
        mix1_kernel<<<dim3((N_NODES * 128 + 255) / 256), dim3(256), 0, stream>>>(
            q, mu_mix, ctx_bf, N_NODES * 128);
        launch_gemm<8, 1, true, false, true>(ctx_bf, Wmix1P[l],
                                             bmix1 + (size_t)l * 128,
                                             nullptr, t2_bf, N_NODES, 128, stream);
        launch_gemm<4, 0, true, true, false>(t2_bf, Wmix2P[l],
                                             bmix2 + (size_t)l * 384,
                                             xc /*=xm*/, nullptr, N_NODES, 384,
                                             stream);
        mix2_kernel<<<dim3((N_NODES * 128 + 255) / 256), dim3(256), 0, stream>>>(
            q, mu, xc, mu_mix, q_bf, N_NODES * 128);
    }
    // q and mu were accumulated directly in d_out — nothing left to copy.
    (void)in_sizes; (void)n_in; (void)out_size; (void)ws_size;
}
